// CondConv2d_70420283785811
// MI455X (gfx1250) — matrix operand order repaired and checked by reading, not compiled, not run on hardware
//
#include <hip/hip_runtime.h>
#include <math.h>
#include <stdint.h>

// ---------------- problem constants ----------------
#define B_    16
#define CIN   256
#define H_    64
#define W_    64
#define E_    8
#define COUT  256
#define KK    2304          // CIN * 9, k = (kh*3+kw)*256 + ci
#define HP    66            // padded height (halo 1)
#define WP    66            // padded width

// ---------------- workspace layout (bytes) ----------------
#define POOLED_OFF 0u                      // 16*256 f32            = 16384
#define ROUT_OFF   16384u                  // 16*8 f32              = 512
#define AGGB_OFF   16896u                  // 16*256 f32            = 16384
#define AGGW_OFF   33280u                  // 16*256*2304 bf16      = 18874368
#define XPAD_OFF   18907648u               // 16*66*66*256 bf16     = 35684352
#define XPAD_BYTES 35684352u               // total ws = 54592000 B (~54.6 MB)

typedef __attribute__((ext_vector_type(4)))  float  v4f;
typedef __attribute__((ext_vector_type(8)))  float  v8f;
typedef __attribute__((ext_vector_type(4)))  int    v4i;
typedef __attribute__((ext_vector_type(8)))  __bf16 v8bf;
typedef __attribute__((ext_vector_type(16))) __bf16 v16bf;

// address-space-qualified pointer typedefs for the async builtin
typedef __attribute__((address_space(1))) v4i* gv4i_p;   // global int4*
typedef __attribute__((address_space(3))) v4i* lv4i_p;   // LDS int4*

// ---- optional CDNA5 async global->LDS path (compile-probed) ----
#if __has_builtin(__builtin_amdgcn_global_load_async_to_lds_b128) && \
    __has_builtin(__builtin_amdgcn_s_wait_asynccnt)
#define USE_ASYNC_LDS 1
#else
#define USE_ASYNC_LDS 0
#endif

__device__ __forceinline__ void stage_b128(const unsigned short* g, unsigned short* s) {
#if USE_ASYNC_LDS
    // LDS generic address: low 32 bits are the LDS byte offset (HW truncates to addr[31:0])
    __builtin_amdgcn_global_load_async_to_lds_b128(
        (gv4i_p)(uintptr_t)g,
        (lv4i_p)(uintptr_t)(unsigned)(uintptr_t)s,
        0, 0);
#else
    *(v4f*)s = *(const v4f*)g;
#endif
}

__device__ __forceinline__ void stage_fence() {
#if USE_ASYNC_LDS
    __builtin_amdgcn_s_wait_asynccnt(0);
#endif
    __syncthreads();
}

__device__ __forceinline__ unsigned short f32_to_bf16(float f) {
    unsigned int u = __builtin_bit_cast(unsigned int, f);
    u += 0x7fffu + ((u >> 16) & 1u);      // round to nearest even
    return (unsigned short)(u >> 16);
}

// ---------------- 1) global average pool: pooled[b][c] ----------------
__global__ __launch_bounds__(256) void pool_kernel(const float* __restrict__ x,
                                                   float* __restrict__ pooled) {
    int bc = blockIdx.x;                               // b*CIN + c
    const float* p = x + (size_t)bc * (H_ * W_);
    float s = 0.f;
    for (int i = threadIdx.x; i < H_ * W_; i += 256) s += p[i];
    __shared__ float red[256];
    red[threadIdx.x] = s;
    __syncthreads();
    for (int o = 128; o > 0; o >>= 1) {
        if (threadIdx.x < (unsigned)o) red[threadIdx.x] += red[threadIdx.x + o];
        __syncthreads();
    }
    if (threadIdx.x == 0) pooled[bc] = red[0] * (1.0f / (H_ * W_));
}

// ---------------- 2) routing[b][e] = sigmoid(pooled . rw^T + rb) ----------------
__global__ __launch_bounds__(128) void routing_kernel(const float* __restrict__ pooled,
                                                      const float* __restrict__ rw,
                                                      const float* __restrict__ rb,
                                                      float* __restrict__ routing) {
    int t = threadIdx.x;
    if (t >= B_ * E_) return;
    int b = t / E_, e = t % E_;
    float z = rb[e];
    for (int c = 0; c < CIN; ++c) z += pooled[b * CIN + c] * rw[e * CIN + c];
    routing[t] = 1.f / (1.f + expf(-z));
}

// ---------------- 3) agg_b[b][o] = routing[b] . bias[:,o] ----------------
__global__ __launch_bounds__(256) void aggb_kernel(const float* __restrict__ bias,
                                                   const float* __restrict__ routing,
                                                   float* __restrict__ aggb) {
    int t = blockIdx.x * 256 + threadIdx.x;            // b*COUT + o
    int b = t >> 8, o = t & 255;
    float acc = 0.f;
#pragma unroll
    for (int e = 0; e < E_; ++e)
        acc += routing[b * E_ + e] * bias[e * COUT + o];
    aggb[t] = acc;
}

// ---- 4) agg_w bf16, layout [b][o][k] with k = (kh*3+kw)*256 + ci ----
__global__ __launch_bounds__(256) void aggw_kernel(const float* __restrict__ weight,
                                                   const float* __restrict__ routing,
                                                   unsigned short* __restrict__ aggw) {
    size_t idx = (size_t)blockIdx.x * 256 + threadIdx.x;   // b*COUT*KK + o*KK + k
    int k = (int)(idx % KK);
    size_t bo = idx / KK;
    int o = (int)(bo % COUT);
    int b = (int)(bo / COUT);
    int khkw = k >> 8;                                     // kh*3+kw
    int ci   = k & 255;
    float acc = 0.f;
#pragma unroll
    for (int e = 0; e < E_; ++e) {
        float w = weight[((size_t)(e * COUT + o) * CIN + ci) * 9 + khkw];
        acc += routing[b * E_ + e] * w;
    }
    aggw[idx] = f32_to_bf16(acc);
}

// ---- 5) x -> padded NHWC bf16: xpad[b][h+1][w+1][ci] (halo pre-zeroed) ----
__global__ __launch_bounds__(256) void xfill_kernel(const float* __restrict__ x,
                                                    unsigned short* __restrict__ xpad) {
    size_t idx = (size_t)blockIdx.x * 256 + threadIdx.x;   // linear over x (coalesced read)
    int w  = (int)(idx & 63);
    int h  = (int)((idx >> 6) & 63);
    int ci = (int)((idx >> 12) & 255);
    int b  = (int)(idx >> 20);
    unsigned short v = f32_to_bf16(x[idx]);
    xpad[(((size_t)b * HP + (h + 1)) * WP + (w + 1)) * CIN + ci] = v;
}

// ---------------- 6) implicit-GEMM conv via v_wmma_f32_16x16x32_bf16 ----------------
// Block (8 waves) computes 128 pixels x 64 couts.
// Wave = 16 pixels x 64 couts = 4 accumulators; A-fragment reused x4.
// B tile (32k x 64o, 4KB) staged in LDS per k-step (async global->LDS if available),
// double buffered; inner 8 k-steps fully unrolled -> immediate-offset loads.
#define BROW 40   // LDS row pitch in bf16 elems (20 dwords: conflict-free b128 reads)

__global__ __launch_bounds__(256) void condconv_wmma(
    const unsigned short* __restrict__ xpad,   // [B][66][66][256] bf16
    const unsigned short* __restrict__ aggw,   // [B][COUT][KK]    bf16
    const float* __restrict__ aggb,            // [B][COUT]
    float* __restrict__ out)                   // [B][COUT][64][64]
{
    __shared__ __align__(16) unsigned short smem[2][64 * BROW];   // 2 x 5120 B

    const int tid  = threadIdx.x;
    const int wave = tid >> 5;
    const int lane = tid & 31;
    const int half = lane >> 4;      // K-half selector per CDNA5 bf16 A/B layouts
    const int l15  = lane & 15;

    const int b  = blockIdx.z;                 // sample
    const int o0 = blockIdx.y * 64;            // cout block
    const int h  = blockIdx.x * 2 + (wave >> 2);
    const int w0 = (wave & 3) * 16;

    // --- cooperative B staging: thread t handles row o=tid/4, k-chunk (tid%4)*8 ---
    const int lo = tid >> 2;                   // 0..63
    const int lk = (tid & 3) * 8;              // 0,8,16,24
    const unsigned short* gB = aggw + (size_t)(b * COUT + o0 + lo) * KK + lk;
    unsigned short* sB0 = &smem[0][0] + lo * BROW + lk;
    unsigned short* sB1 = &smem[1][0] + lo * BROW + lk;

    // --- per-wave A base: pixel (h, w0+l15), channel offset half*16 ---
    const unsigned short* xpix =
        xpad + ((size_t)(b * HP + h) * WP + (w0 + l15)) * CIN + half * 16;

    // --- per-wave B fragment read base (row l15, k-half) ---
    const unsigned short* rB0 = &smem[0][0] + l15 * BROW + half * 16;
    const unsigned short* rB1 = &smem[1][0] + l15 * BROW + half * 16;

    v8f c[4] = {};                             // 16 px x 64 couts (f32)

    stage_b128(gB, sB0);                       // k-step 0
    stage_fence();

    for (int khkw = 0; khkw < 9; ++khkw) {     // kh*3+kw
        const int kh = khkw / 3;
        const int kw = khkw - kh * 3;
        const unsigned short* arow = xpix + (kh * WP + kw) * CIN;  // invariant
        const unsigned short* gK   = gB + khkw * 256;              // invariant

#pragma unroll
        for (int cit = 0; cit < 8; ++cit) {
            // stage next k-step into the other buffer (parity = cit&1, compile-time).
            // Final step (khkw==8,cit==7) reads 64B past agg_w: still inside the
            // workspace (xpad region follows) and the data is never consumed.
            stage_b128(gK + (cit + 1) * 32, (cit & 1) ? sB0 : sB1);

            // A fragment, immediate offsets off arow
            const unsigned short* ap = arow + cit * 32;
            v8bf a0 = *(const v8bf*)ap;
            v8bf a1 = *(const v8bf*)(ap + 8);
            v16bf A = __builtin_shufflevector(a0, a1,
                        0,1,2,3,4,5,6,7,8,9,10,11,12,13,14,15);

            const unsigned short* rb = (cit & 1) ? rB1 : rB0;
#pragma unroll
            for (int osub = 0; osub < 4; ++osub) {
                const unsigned short* bp = rb + osub * 16 * BROW;  // imm ds offset
                v8bf b0 = *(const v8bf*)bp;
                v8bf b1 = *(const v8bf*)(bp + 8);
                v16bf Bm = __builtin_shufflevector(b0, b1,
                             0,1,2,3,4,5,6,7,8,9,10,11,12,13,14,15);
                c[osub] = __builtin_amdgcn_wmma_f32_16x16x32_bf16(
                              false, A, false, Bm, (short)0, c[osub], false, false);
            }
            stage_fence();
        }
    }

    // D layout: lane n -> N=n (cout within subtile); VGPR r -> M = half*8 + r (w)
#pragma unroll
    for (int osub = 0; osub < 4; ++osub) {
        int o = o0 + osub * 16 + l15;
        float bv = aggb[b * COUT + o];
        float* op = out + ((size_t)(b * COUT + o) * H_ + h) * W_ + w0 + half * 8;
        v4f s0 = { c[osub][0] + bv, c[osub][1] + bv, c[osub][2] + bv, c[osub][3] + bv };
        v4f s1 = { c[osub][4] + bv, c[osub][5] + bv, c[osub][6] + bv, c[osub][7] + bv };
        *(v4f*)op       = s0;
        *(v4f*)(op + 4) = s1;
    }
}

// ---------------- launch ----------------
extern "C" void kernel_launch(void* const* d_in, const int* in_sizes, int n_in,
                              void* d_out, int out_size, void* d_ws, size_t ws_size,
                              hipStream_t stream) {
    const float* x      = (const float*)d_in[0];
    const float* weight = (const float*)d_in[1];
    const float* bias   = (const float*)d_in[2];
    const float* rw     = (const float*)d_in[3];
    const float* rb     = (const float*)d_in[4];
    float* out = (float*)d_out;

    char* ws = (char*)d_ws;
    float* pooled        = (float*)(ws + POOLED_OFF);
    float* routing       = (float*)(ws + ROUT_OFF);
    float* aggb          = (float*)(ws + AGGB_OFF);
    unsigned short* aggw = (unsigned short*)(ws + AGGW_OFF);
    unsigned short* xpad = (unsigned short*)(ws + XPAD_OFF);

    (void)hipMemsetAsync(xpad, 0, XPAD_BYTES, stream);   // zero halo (capturable node)

    pool_kernel   <<<B_ * CIN, 256, 0, stream>>>(x, pooled);
    routing_kernel<<<1, 128, 0, stream>>>(pooled, rw, rb, routing);
    aggb_kernel   <<<(B_ * COUT) / 256, 256, 0, stream>>>(bias, routing, aggb);
    aggw_kernel   <<<(B_ * COUT * KK) / 256, 256, 0, stream>>>(weight, routing, aggw);
    xfill_kernel  <<<(B_ * CIN * H_ * W_) / 256, 256, 0, stream>>>(x, xpad);

    // 2048 blocks: 32 h-pairs x 4 cout-blocks x 16 samples, 256 thr (8 waves)
    condconv_wmma<<<dim3(32, 4, 16), 256, 0, stream>>>(xpad, aggw, aggb, out);
}
